// EnhancedDGAR_63977832841454
// MI455X (gfx1250) — matrix operand order repaired and checked by reading, not compiled, not run on hardware
//
#include <hip/hip_runtime.h>

// ---------------------------------------------------------------------------
// EnhancedDGAR forward for MI455X (gfx1250, wave32, WMMA f32<-f16)
// f16 operand buffers + LDS-staged, register-tiled WMMA GEMM
// ---------------------------------------------------------------------------
#define NND 8192    // nodes
#define BB  512     // graphs
#define FD  64      // node features
#define HID 128
#define WD  32      // window
#define TT  5       // time steps
#define EE  122880  // edges (512 * 240)
#define RR  4096    // B * 8 rows
#define BNI 0.9999950000374997f   // 1/sqrt(1+1e-5)

typedef __attribute__((ext_vector_type(16))) _Float16 v16h;
typedef __attribute__((ext_vector_type(8)))  _Float16 v8h;
typedef __attribute__((ext_vector_type(8)))  float    v8f;

__device__ __constant__ int d_meas[8]    = {0,3,5,6,8,11,13,14};
__device__ __constant__ int d_unm[8]     = {1,2,4,7,9,10,12,15};
__device__ __constant__ int d_ismeas[16] = {1,0,0,1,0,1,1,0,1,0,0,1,0,1,1,0};
__device__ __constant__ int d_pos[16]    = {0,0,1,1,2,2,3,3,4,4,5,5,6,6,7,7};

__device__ __forceinline__ float sigmf(float x){ return 1.f/(1.f+expf(-x)); }

// f32 -> f16 conversion (weights / activations staged once)
__global__ void k_cvt(const float* __restrict__ s, _Float16* __restrict__ d, int n)
{
    int i = blockIdx.x * 256 + threadIdx.x;
    if (i < n) d[i] = (_Float16)s[i];
}

// ---------------------------------------------------------------------------
// k_gemm2: C = A @ W^T (+bias)(+relu). A: MxK f16 (lda), W: NxK f16 (ldw).
// 128 threads = 4 waves; block tile 64(M)x64(N); wave tile 16x64 (4 wmma acc).
// B panel (64x32 f16) staged via LDS. M%64==0, N%64==0, K%32==0.
// C (f32) and Ch (f16 mirror) are each optional.
// ---------------------------------------------------------------------------
__global__ void k_gemm2(const _Float16* __restrict__ Ah, const _Float16* __restrict__ Wh,
                        float* __restrict__ C, _Float16* __restrict__ Ch,
                        const float* __restrict__ bias,
                        int K, int lda, int ldw, int ldc, int relu)
{
    __shared__ __align__(16) _Float16 Bs[64 * 32];
    const int tid  = threadIdx.x;
    const int w    = tid >> 5;          // wave 0..3
    const int lane = tid & 31;
    const int half = lane >> 4;
    const int l16  = lane & 15;

    const int arow = blockIdx.y * 64 + w * 16 + l16;
    const _Float16* Arow = Ah + (size_t)arow * lda;

    // cooperative B staging: thread t -> row tid/2, half-panel (tid&1)*16
    const int brow   = tid >> 1;
    const int bcol16 = (tid & 1) * 16;
    const _Float16* Bsrc = Wh + (size_t)(blockIdx.x * 64 + brow) * ldw + bcol16;

    v8f acc[4] = {};
    for (int k0 = 0; k0 < K; k0 += 32) {
        if (k0 + 32 < K) {
            __builtin_prefetch(Arow + k0 + 32, 0, 1);
            __builtin_prefetch(Bsrc + k0 + 32, 0, 1);
        }
        v8h b0 = *(const v8h*)(Bsrc + k0);
        v8h b1 = *(const v8h*)(Bsrc + k0 + 8);
        __syncthreads();                              // previous-iter reads done
        *(v8h*)(&Bs[brow * 32 + bcol16])     = b0;    // ds_store_b128
        *(v8h*)(&Bs[brow * 32 + bcol16 + 8]) = b1;
        __syncthreads();                              // panel visible

        // A fragment (contiguous k order: k = 16*half + s, matches B)
        v8h alo = *(const v8h*)(Arow + k0 + half * 16);
        v8h ahi = *(const v8h*)(Arow + k0 + half * 16 + 8);
        v16h a;
#pragma unroll
        for (int j = 0; j < 8; ++j) { a[j] = alo[j]; a[j + 8] = ahi[j]; }

#pragma unroll
        for (int n = 0; n < 4; ++n) {
            const _Float16* bp = &Bs[(n * 16 + l16) * 32 + half * 16];
            v8h blo = *(const v8h*)bp;                // ds_load_b128
            v8h bhi = *(const v8h*)(bp + 8);
            v16h b;
#pragma unroll
            for (int j = 0; j < 8; ++j) { b[j] = blo[j]; b[j + 8] = bhi[j]; }
            acc[n] = __builtin_amdgcn_wmma_f32_16x16x32_f16(false, a, false, b,
                                                            (short)0, acc[n], false, false);
        }
    }

    const int crow0 = blockIdx.y * 64 + w * 16 + half * 8;
#pragma unroll
    for (int n = 0; n < 4; ++n) {
        const int ccol = blockIdx.x * 64 + n * 16 + l16;
        const float bv = bias ? bias[ccol] : 0.f;
#pragma unroll
        for (int r = 0; r < 8; ++r) {
            float v = acc[n][r] + bv;
            if (relu) v = fmaxf(v, 0.f);
            const size_t off = (size_t)(crow0 + r) * ldc + ccol;
            if (C)  C[off]  = v;
            if (Ch) Ch[off] = (_Float16)v;
        }
    }
}

// ---------------------------------------------------------------------------
// k_gemm (v1, f32 operands, one wave per 16x16 tile) — kept for N=32 out2 GEMM
// ---------------------------------------------------------------------------
__global__ void k_gemm(const float* __restrict__ A, const float* __restrict__ Wm,
                       float* __restrict__ C, const float* __restrict__ bias,
                       int K, int lda, int ldw, int ldc, int relu)
{
    const int lane = threadIdx.x;
    const int half = lane >> 4;
    const int l16  = lane & 15;
    const float* Ar = A  + (size_t)(blockIdx.y * 16 + l16) * lda;
    const float* Wr = Wm + (size_t)(blockIdx.x * 16 + l16) * ldw;

    v8f acc = {};
    for (int k0 = 0; k0 < K; k0 += 32) {
        v16h a, b;
        const int ka = k0 + half * 16;
        const int kb = k0 + half * 16;
#pragma unroll
        for (int j = 0; j < 8; ++j) {
            a[j]     = (_Float16)Ar[ka + j];
            a[j + 8] = (_Float16)Ar[ka + 8 + j];
        }
#pragma unroll
        for (int j = 0; j < 16; ++j) b[j] = (_Float16)Wr[kb + j];
        acc = __builtin_amdgcn_wmma_f32_16x16x32_f16(false, a, false, b,
                                                     (short)0, acc, false, false);
    }
    const int crow0 = blockIdx.y * 16 + half * 8;
    const int ccol  = blockIdx.x * 16 + l16;
    const float bv = bias ? bias[ccol] : 0.f;
#pragma unroll
    for (int r = 0; r < 8; ++r) {
        float v = acc[r] + bv;
        if (relu) v = fmaxf(v, 0.f);
        C[(size_t)(crow0 + r) * ldc + ccol] = v;
    }
}

// ---------------------------------------------------------------------------
// GAT helpers
// ---------------------------------------------------------------------------
__global__ void k_we(const float* __restrict__ geW, const float* __restrict__ atte,
                     float* __restrict__ we)
{
    int t = threadIdx.x;
    if (t < 64) {
        int h = t >> 4, c = t & 15;
        float acc = 0.f;
        for (int d = 0; d < 128; ++d) acc += geW[(h*128 + d)*16 + c] * atte[h*128 + d];
        we[h*16 + c] = acc;
    }
}

__global__ void k_asrcdst(const float* __restrict__ xl, const float* __restrict__ atts,
                          const float* __restrict__ attd, float* __restrict__ asrc,
                          float* __restrict__ adst)
{
    __shared__ float r1[128], r2[128];
    int n = blockIdx.x, d = threadIdx.x;
    for (int h = 0; h < 4; ++h) {
        float v = xl[(size_t)n*512 + h*128 + d];
        r1[d] = v * atts[h*128 + d];
        r2[d] = v * attd[h*128 + d];
        __syncthreads();
        for (int s = 64; s > 0; s >>= 1) {
            if (d < s) { r1[d] += r1[d+s]; r2[d] += r2[d+s]; }
            __syncthreads();
        }
        if (d == 0) { asrc[n*4 + h] = r1[0]; adst[n*4 + h] = r2[0]; }
        __syncthreads();
    }
}

__global__ void k_aedge(const float* __restrict__ ea, const float* __restrict__ we,
                        float* __restrict__ aedge)
{
    int e = blockIdx.x * 256 + threadIdx.x;
    if (e >= EE) return;
    for (int h = 0; h < 4; ++h) {
        float acc = 0.f;
        for (int c = 0; c < 16; ++c) acc += ea[(size_t)e*16 + c] * we[h*16 + c];
        aedge[(size_t)e*4 + h] = acc;
    }
}

// softmax over 15 in-edges + aggregation + bias; writes f16 (feeds proj GEMM)
__global__ void k_gatagg(const float* __restrict__ xl, const float* __restrict__ asrc,
                         const float* __restrict__ adst, const float* __restrict__ aedge,
                         const int* __restrict__ ei, const float* __restrict__ gbias,
                         _Float16* __restrict__ gnnh)
{
    __shared__ float lg[15][4];
    __shared__ float al[15][4];
    __shared__ int   srcs[15];
    int n = blockIdx.x, b = n >> 4, i = n & 15, tid = threadIdx.x;
    if (tid < 15) srcs[tid] = ei[b*240 + i*15 + tid];
    __syncthreads();
    if (tid < 60) {
        int m = tid >> 2, h = tid & 3;
        int e = b*240 + i*15 + m;
        float v = asrc[(size_t)srcs[m]*4 + h] + adst[(size_t)n*4 + h] + aedge[(size_t)e*4 + h];
        lg[m][h] = v > 0.f ? v : 0.2f * v;
    }
    __syncthreads();
    if (tid < 4) {
        float mx = -1e30f;
        for (int m = 0; m < 15; ++m) mx = fmaxf(mx, lg[m][tid]);
        float den = 0.f;
        for (int m = 0; m < 15; ++m) { float e = expf(lg[m][tid]-mx); al[m][tid] = e; den += e; }
        float inv = 1.f / den;
        for (int m = 0; m < 15; ++m) al[m][tid] *= inv;
    }
    __syncthreads();
    int d = tid;
    for (int h = 0; h < 4; ++h) {
        float acc = 0.f;
        for (int m = 0; m < 15; ++m) acc += al[m][h] * xl[(size_t)srcs[m]*512 + h*128 + d];
        gnnh[(size_t)n*512 + h*128 + d] = (_Float16)(acc + gbias[h*128 + d]);
    }
}

// LayerNorm(HID) + affine + relu (f32 in/out). param index = row / pdiv
__global__ void k_ln(const float* __restrict__ X, float* __restrict__ Y,
                     const float* __restrict__ g, const float* __restrict__ b, int pdiv)
{
    __shared__ float red[128];
    int row = blockIdx.x, j = threadIdx.x;
    float v = X[(size_t)row*HID + j];
    red[j] = v; __syncthreads();
    for (int s = 64; s > 0; s >>= 1) { if (j < s) red[j] += red[j+s]; __syncthreads(); }
    float mean = red[0] * (1.f/HID);
    __syncthreads();
    float dv = v - mean;
    red[j] = dv*dv; __syncthreads();
    for (int s = 64; s > 0; s >>= 1) { if (j < s) red[j] += red[j+s]; __syncthreads(); }
    float var = red[0] * (1.f/HID);
    int p = (row / pdiv) * HID + j;
    float out = dv * rsqrtf(var + 1e-5f) * g[p] + b[p];
    Y[(size_t)row*HID + j] = fmaxf(out, 0.f);
}

// ---------------------------------------------------------------------------
// GRU
// ---------------------------------------------------------------------------
__global__ void k_gather_hist(const float* __restrict__ hist, _Float16* __restrict__ X0h)
{
    int idx = blockIdx.x * 256 + threadIdx.x;           // TT*RR*WD total
    int t = idx / (RR*WD); int rem = idx % (RR*WD);
    int r = rem / WD; int w = rem % WD;
    int b = r >> 3, mi = r & 7;
    int node = b*16 + d_meas[mi];
    X0h[idx] = (_Float16)hist[(size_t)node*TT*WD + t*WD + w];
}

// fused GRU gates; h kept f32 + f16 mirror (GEMM operand); Y written f16
__global__ void k_grugate(const float* __restrict__ gi, const float* __restrict__ gh,
                          float* __restrict__ h, _Float16* __restrict__ hh,
                          _Float16* __restrict__ Y, int gi_row0, int t_store, int back)
{
    int r = blockIdx.x, j = threadIdx.x;
    const float* gir = gi + (size_t)(gi_row0 + r) * 384;
    const float* ghr = gh + (size_t)r * 384;
    float rg = sigmf(gir[j]       + ghr[j]);
    float z  = sigmf(gir[128 + j] + ghr[128 + j]);
    float nn = tanhf(gir[256 + j] + rg * ghr[256 + j]);
    float hp = h[(size_t)r*HID + j];
    float hn = (1.f - z) * nn + z * hp;
    h[(size_t)r*HID + j]  = hn;
    hh[(size_t)r*HID + j] = (_Float16)hn;
    Y[((size_t)t_store*RR + r)*256 + back*128 + j] = (_Float16)hn;
}

// ---------------------------------------------------------------------------
// Cross attention
// ---------------------------------------------------------------------------
__global__ void k_gather_u(const float* __restrict__ gnnf, float* __restrict__ umat,
                           _Float16* __restrict__ umath)
{
    int idx = blockIdx.x * 256 + threadIdx.x;           // RR*HID total
    int r = idx / HID, d = idx % HID;
    int b = r >> 3, qi = r & 7;
    float v = gnnf[(size_t)(b*16 + d_unm[qi])*HID + d];
    umat[idx]  = v;
    umath[idx] = (_Float16)v;
}

__global__ void k_attn(const float* __restrict__ qb, const float* __restrict__ kb,
                       const float* __restrict__ vb, const float* __restrict__ umat,
                       _Float16* __restrict__ combh)
{
    __shared__ float sc[8][8];
    int b = blockIdx.x, t = threadIdx.x;                // 64 threads
    {
        int qi = t >> 3, ki = t & 7;
        const float* qr = qb + (size_t)(b*8 + qi)*HID;
        const float* kr = kb + (size_t)(b*8 + ki)*HID;
        float acc = 0.f;
        for (int d = 0; d < HID; ++d) acc += qr[d]*kr[d];
        sc[qi][ki] = acc * 0.08838834764831845f;        // 1/sqrt(128)
    }
    __syncthreads();
    if (t < 8) {
        float mx = -1e30f;
        for (int k = 0; k < 8; ++k) mx = fmaxf(mx, sc[t][k]);
        float den = 0.f;
        for (int k = 0; k < 8; ++k) { float e = expf(sc[t][k]-mx); sc[t][k] = e; den += e; }
        float inv = 1.f/den;
        for (int k = 0; k < 8; ++k) sc[t][k] *= inv;
    }
    __syncthreads();
    for (int idx = t; idx < 8*HID; idx += 64) {
        int qi = idx / HID, d = idx % HID;
        float acc = umat[(size_t)(b*8 + qi)*HID + d];
        for (int k = 0; k < 8; ++k) acc += sc[qi][k] * vb[(size_t)(b*8 + k)*HID + d];
        combh[((size_t)qi*BB + b)*HID + d] = (_Float16)acc;   // [u][b][d]
    }
}

// ---------------------------------------------------------------------------
// TCN (broadcast-input conv collapses to 3/5 distinct positions)
// ---------------------------------------------------------------------------
__global__ void k_wprep(const float* __restrict__ t1W, const float* __restrict__ t2W,
                        _Float16* __restrict__ Wc1, _Float16* __restrict__ W2k)
{
    int idx = blockIdx.x * 256 + threadIdx.x;           // 8*3*256*128 total
    {
        int u = idx / (3*256*128); int rem = idx % (3*256*128);
        int c = rem / (256*128);   int oi = rem % (256*128);
        int o = oi / 128, i2 = oi % 128;
        const float* wp = t1W + ((size_t)(u*256 + o)*128 + i2)*3;
        float v = (c == 0) ? (wp[1] + wp[2]) : (c == 1) ? (wp[0] + wp[1] + wp[2]) : (wp[0] + wp[1]);
        Wc1[idx] = (_Float16)v;
    }
    {
        int u = idx / (3*128*256); int rem = idx % (3*128*256);
        int k = rem / (128*256);   int oi = rem % (128*256);
        int o = oi / 256, i2 = oi % 256;
        W2k[idx] = (_Float16)t2W[((size_t)(u*128 + o)*256 + i2)*3 + k];
    }
}

// y1h = f16(relu((conv1 + b1)*BNI*g1 + be1)) over [8][3][512][256]
__global__ void k_bn1(const float* __restrict__ y1, _Float16* __restrict__ y1h,
                      const float* __restrict__ b1, const float* __restrict__ g1,
                      const float* __restrict__ be1)
{
    size_t idx = (size_t)blockIdx.x * 256 + threadIdx.x;  // 8*3*512*256 total
    int u = (int)(idx / ((size_t)3*512*256));
    int o = (int)(idx & 255);
    int p = u*256 + o;
    float v = (y1[idx] + b1[p]) * BNI * g1[p] + be1[p];
    y1h[idx] = (_Float16)fmaxf(v, 0.f);
}

__global__ void k_combine(const float* __restrict__ P, const float* __restrict__ b2,
                          const float* __restrict__ g2, const float* __restrict__ be2,
                          _Float16* __restrict__ pmh)
{
    int idx = blockIdx.x * 256 + threadIdx.x;           // 8*512*128 total
    int u = idx / (512*128); int rem = idx % (512*128);
    int row = rem / 128; int o = rem % 128;
    int pp = u*128 + o;
    float cb = b2[pp], cg = g2[pp], ce = be2[pp];
    const size_t base = (size_t)u*9*512*128 + (size_t)row*128 + o;
#define PCK(c,k) P[base + (size_t)((c)*3 + (k))*512*128]
    float pa0 = PCK(0,0), pa1 = PCK(0,1);
    float pb0 = PCK(1,0), pb1 = PCK(1,1), pb2 = PCK(1,2);
    float pc1 = PCK(2,1), pc2 = PCK(2,2);
#undef PCK
    auto bn = [&](float x){ float v = (x + cb) * BNI * cg + ce; return fmaxf(v, 0.f); };
    float v0   = bn(pa1 + pb2);
    float v1   = bn(pa0 + pb1 + pb2);
    float vm   = bn(pb0 + pb1 + pb2);
    float v126 = bn(pb0 + pb1 + pc2);
    float v127 = bn(pb0 + pc1);
    pmh[idx] = (_Float16)((v0 + v1 + 124.f*vm + v126 + v127) * (1.f/128.f));
}

__global__ void k_scatter(const float* __restrict__ target, const float* __restrict__ preds,
                          float* __restrict__ out)
{
    int idx = blockIdx.x * 256 + threadIdx.x;           // N*WD total
    int n = idx / WD, w = idx % WD;
    int i = n & 15, b = n >> 4;
    if (d_ismeas[i]) out[idx] = target[idx];
    else             out[idx] = preds[((size_t)d_pos[i]*BB + b)*WD + w];
}

// ---------------------------------------------------------------------------
extern "C" void kernel_launch(void* const* d_in, const int* in_sizes, int n_in,
                              void* d_out, int out_size, void* d_ws, size_t ws_size,
                              hipStream_t stream)
{
    (void)in_sizes; (void)n_in; (void)out_size; (void)ws_size;
    const float* x        = (const float*)d_in[0];
    const float* edge_attr= (const float*)d_in[1];
    const float* history  = (const float*)d_in[2];
    const float* target   = (const float*)d_in[3];
    const float* emb_W    = (const float*)d_in[4];
    const float* emb_b    = (const float*)d_in[5];
    const float* gat_W    = (const float*)d_in[6];
    const float* att_src  = (const float*)d_in[7];
    const float* att_dst  = (const float*)d_in[8];
    const float* gedge_W  = (const float*)d_in[9];
    const float* att_edge = (const float*)d_in[10];
    const float* gat_bias = (const float*)d_in[11];
    const float* proj_W   = (const float*)d_in[12];
    const float* proj_b   = (const float*)d_in[13];
    const float* pln_g    = (const float*)d_in[14];
    const float* pln_b    = (const float*)d_in[15];
    const float* g0_Wih   = (const float*)d_in[16];
    const float* g0_Whh   = (const float*)d_in[17];
    const float* g0_bih   = (const float*)d_in[18];
    const float* g0_bhh   = (const float*)d_in[19];
    const float* g1_Wih   = (const float*)d_in[20];
    const float* g1_Whh   = (const float*)d_in[21];
    const float* g1_bih   = (const float*)d_in[22];
    const float* g1_bhh   = (const float*)d_in[23];
    const float* hist_W   = (const float*)d_in[24];
    const float* hist_b   = (const float*)d_in[25];
    const float* q_W      = (const float*)d_in[26];
    const float* q_b      = (const float*)d_in[27];
    const float* kk_W     = (const float*)d_in[28];
    const float* kk_b     = (const float*)d_in[29];
    const float* v_W      = (const float*)d_in[30];
    const float* v_b      = (const float*)d_in[31];
    const float* t1_W     = (const float*)d_in[32];
    const float* t1_b     = (const float*)d_in[33];
    const float* bn1_g    = (const float*)d_in[34];
    const float* bn1_b    = (const float*)d_in[35];
    const float* t2_W     = (const float*)d_in[36];
    const float* t2_b     = (const float*)d_in[37];
    const float* bn2_g    = (const float*)d_in[38];
    const float* bn2_b    = (const float*)d_in[39];
    const float* o1_W     = (const float*)d_in[40];
    const float* o1_b     = (const float*)d_in[41];
    const float* oln_g    = (const float*)d_in[42];
    const float* oln_b    = (const float*)d_in[43];
    const float* o2_W     = (const float*)d_in[44];
    const float* o2_b     = (const float*)d_in[45];
    const int*   ei       = (const int*)d_in[46];      // [2,E] int32
    float* out = (float*)d_out;

    // ---- workspace (byte bump allocator, 256B aligned blocks) ----
    char* base = (char*)d_ws;
    size_t off = 0;
    auto alloc = [&](size_t bytes) -> void* {
        void* r = base + off; off += (bytes + 255) & ~(size_t)255; return r;
    };
    auto f32buf = [&](size_t n) -> float*     { return (float*)alloc(n * 4); };
    auto f16buf = [&](size_t n) -> _Float16*  { return (_Float16*)alloc(n * 2); };

    // f32 intermediates
    float* xl   = f32buf((size_t)NND*512);
    float* asrc = f32buf((size_t)NND*4);
    float* adst = f32buf((size_t)NND*4);
    float* wev  = f32buf(64);
    float* aedg = f32buf((size_t)EE*4);
    float* ptmp = f32buf((size_t)NND*HID);
    float* gnnf = f32buf((size_t)NND*HID);
    float* giF  = f32buf((size_t)TT*RR*384);
    float* giB  = f32buf((size_t)TT*RR*384);
    float* ghb  = f32buf((size_t)RR*384);
    float* hb   = f32buf((size_t)RR*HID);
    float* umat = f32buf((size_t)RR*HID);
    float* qv   = f32buf((size_t)RR*HID);
    float* kv   = f32buf((size_t)RR*HID);
    float* vv   = f32buf((size_t)RR*HID);
    float* y1b  = f32buf((size_t)8*3*512*256);
    float* Pb   = f32buf((size_t)8*9*512*128);
    float* z1   = f32buf((size_t)8*512*128);
    float* zb   = f32buf((size_t)8*512*128);
    float* prd  = f32buf((size_t)8*512*32);
    // f16 operand buffers
    _Float16* xh    = f16buf((size_t)NND*FD);
    _Float16* h0h   = f16buf((size_t)NND*HID);
    _Float16* gnnh  = f16buf((size_t)NND*512);
    _Float16* X0h   = f16buf((size_t)TT*RR*WD);
    _Float16* hh    = f16buf((size_t)RR*HID);
    _Float16* Y0h   = f16buf((size_t)TT*RR*256);
    _Float16* Y1h   = f16buf((size_t)TT*RR*256);
    _Float16* hembh = f16buf((size_t)RR*HID);
    _Float16* umath = f16buf((size_t)RR*HID);
    _Float16* combh = f16buf((size_t)8*BB*HID);
    _Float16* Wc1h  = f16buf((size_t)8*3*256*128);
    _Float16* W2kh  = f16buf((size_t)8*3*128*256);
    _Float16* y1h   = f16buf((size_t)8*3*512*256);
    _Float16* pmh   = f16buf((size_t)8*512*128);
    // f16 weights
    _Float16* embWh  = f16buf(128*64);
    _Float16* gatWh  = f16buf(512*128);
    _Float16* projWh = f16buf(128*512);
    _Float16* g0WihH = f16buf(2*384*32);
    _Float16* g0WhhH = f16buf(2*384*128);
    _Float16* g1WihH = f16buf(2*384*256);
    _Float16* g1WhhH = f16buf(2*384*128);
    _Float16* histWh = f16buf(128*256);
    _Float16* qWh    = f16buf(128*128);
    _Float16* kWh    = f16buf(128*128);
    _Float16* vWh    = f16buf(128*128);
    _Float16* o1Wh   = f16buf(8*128*128);

    auto cvt = [&](const float* s, _Float16* d, int n){
        k_cvt<<<(n + 255)/256, 256, 0, stream>>>(s, d, n);
    };
    auto gemm2 = [&](const _Float16* A, const _Float16* Wm, float* C, _Float16* Ch,
                     const float* bias, int M, int Ncols, int K,
                     int lda, int ldw, int ldc, int relu){
        dim3 g(Ncols/64, M/64);
        k_gemm2<<<g, dim3(128), 0, stream>>>(A, Wm, C, Ch, bias, K, lda, ldw, ldc, relu);
    };

    // ---- 0) stage f16 operands ----
    cvt(x, xh, NND*FD);
    cvt(emb_W, embWh, 128*64);
    cvt(gat_W, gatWh, 512*128);
    cvt(proj_W, projWh, 128*512);
    cvt(g0_Wih, g0WihH, 2*384*32);
    cvt(g0_Whh, g0WhhH, 2*384*128);
    cvt(g1_Wih, g1WihH, 2*384*256);
    cvt(g1_Whh, g1WhhH, 2*384*128);
    cvt(hist_W, histWh, 128*256);
    cvt(q_W, qWh, 128*128);
    cvt(kk_W, kWh, 128*128);
    cvt(v_W, vWh, 128*128);
    cvt(o1_W, o1Wh, 8*128*128);

    // ---- 1) embedding + GAT ----
    gemm2(xh, embWh, nullptr, h0h, emb_b, NND, HID, FD, FD, FD, HID, 1);
    gemm2(h0h, gatWh, xl, nullptr, nullptr, NND, 512, HID, HID, HID, 512, 0);
    k_we<<<1, 64, 0, stream>>>(gedge_W, att_edge, wev);
    k_asrcdst<<<NND, 128, 0, stream>>>(xl, att_src, att_dst, asrc, adst);
    k_aedge<<<EE/256, 256, 0, stream>>>(edge_attr, wev, aedg);
    k_gatagg<<<NND, 128, 0, stream>>>(xl, asrc, adst, aedg, ei, gat_bias, gnnh);
    gemm2(gnnh, projWh, ptmp, nullptr, proj_b, NND, HID, 512, 512, 512, HID, 0);
    k_ln<<<NND, 128, 0, stream>>>(ptmp, gnnf, pln_g, pln_b, NND);

    // ---- 2) 2-layer BiGRU ----
    k_gather_hist<<<(TT*RR*WD)/256, 256, 0, stream>>>(history, X0h);
    gemm2(X0h, g0WihH,          giF, nullptr, g0_bih,       TT*RR, 384, WD, WD, WD, 384, 0);
    gemm2(X0h, g0WihH + 384*WD, giB, nullptr, g0_bih + 384, TT*RR, 384, WD, WD, WD, 384, 0);
    for (int dir = 0; dir < 2; ++dir) {
        const float* gi = dir ? giB : giF;
        hipMemsetAsync(hb, 0, (size_t)RR*HID*sizeof(float), stream);
        hipMemsetAsync(hh, 0, (size_t)RR*HID*sizeof(_Float16), stream);
        for (int s = 0; s < TT; ++s) {
            int t = dir ? (TT-1-s) : s;
            gemm2(hh, g0WhhH + (size_t)dir*384*HID, ghb, nullptr, g0_bhh + dir*384,
                  RR, 384, HID, HID, HID, 384, 0);
            k_grugate<<<RR, 128, 0, stream>>>(gi, ghb, hb, hh, Y0h, t*RR, t, dir);
        }
    }
    gemm2(Y0h, g1WihH,           giF, nullptr, g1_bih,       TT*RR, 384, 256, 256, 256, 384, 0);
    gemm2(Y0h, g1WihH + 384*256, giB, nullptr, g1_bih + 384, TT*RR, 384, 256, 256, 256, 384, 0);
    for (int dir = 0; dir < 2; ++dir) {
        const float* gi = dir ? giB : giF;
        hipMemsetAsync(hb, 0, (size_t)RR*HID*sizeof(float), stream);
        hipMemsetAsync(hh, 0, (size_t)RR*HID*sizeof(_Float16), stream);
        for (int s = 0; s < TT; ++s) {
            int t = dir ? (TT-1-s) : s;
            gemm2(hh, g1WhhH + (size_t)dir*384*HID, ghb, nullptr, g1_bhh + dir*384,
                  RR, 384, HID, HID, HID, 384, 0);
            k_grugate<<<RR, 128, 0, stream>>>(gi, ghb, hb, hh, Y1h, t*RR, t, dir);
        }
    }
    gemm2(Y1h + (size_t)(TT-1)*RR*256, histWh, nullptr, hembh, hist_b,
          RR, HID, 256, 256, 256, HID, 0);

    // ---- 3) cross attention ----
    k_gather_u<<<(RR*HID)/256, 256, 0, stream>>>(gnnf, umat, umath);
    gemm2(umath, qWh, qv, nullptr, q_b,  RR, HID, HID, HID, HID, HID, 0);
    gemm2(hembh, kWh, kv, nullptr, kk_b, RR, HID, HID, HID, HID, HID, 0);
    gemm2(hembh, vWh, vv, nullptr, v_b,  RR, HID, HID, HID, HID, HID, 0);
    k_attn<<<BB, 64, 0, stream>>>(qv, kv, vv, umat, combh);

    // ---- 4) TCN heads (broadcast input => 3/5 distinct conv columns) ----
    k_wprep<<<(8*3*256*128)/256, 256, 0, stream>>>(t1_W, t2_W, Wc1h, W2kh);
    for (int u = 0; u < 8; ++u)
        for (int c = 0; c < 3; ++c)
            gemm2(combh + (size_t)u*512*128, Wc1h + (size_t)(u*3+c)*256*128,
                  y1b + (size_t)(u*3+c)*512*256, nullptr, nullptr,
                  512, 256, 128, 128, 128, 256, 0);
    k_bn1<<<(8*3*512*256)/256, 256, 0, stream>>>(y1b, y1h, t1_b, bn1_g, bn1_b);
    for (int u = 0; u < 8; ++u)
        for (int c = 0; c < 3; ++c)
            for (int k = 0; k < 3; ++k) {
                if ((c == 0 && k == 2) || (c == 2 && k == 0)) continue;
                gemm2(y1h + (size_t)(u*3+c)*512*256, W2kh + (size_t)(u*3+k)*128*256,
                      Pb + (size_t)(u*9 + c*3 + k)*512*128, nullptr, nullptr,
                      512, 128, 256, 256, 256, 128, 0);
            }
    k_combine<<<(8*512*128)/256, 256, 0, stream>>>(Pb, t2_b, bn2_g, bn2_b, pmh);

    // ---- 5) output MLPs + scatter ----
    for (int u = 0; u < 8; ++u)
        gemm2(pmh + (size_t)u*512*128, o1Wh + (size_t)u*128*128,
              z1 + (size_t)u*512*128, nullptr, o1_b + u*128,
              512, 128, 128, 128, 128, 128, 0);
    k_ln<<<8*512, 128, 0, stream>>>(z1, zb, oln_g, oln_b, 512);
    for (int u = 0; u < 8; ++u)
        k_gemm<<<dim3(WD/16, 512/16), dim3(32), 0, stream>>>(
            zb + (size_t)u*512*128, o2_W + (size_t)u*32*128,
            prd + (size_t)u*512*32, o2_b + u*32, 128, 128, 128, 32, 0);
    k_scatter<<<(NND*WD)/256, 256, 0, stream>>>(target, prd, out);
}
